// biLM_39453569581724
// MI455X (gfx1250) — matrix operand order repaired and checked by reading, not compiled, not run on hardware
//
#include <hip/hip_runtime.h>

#define T_LEN 512
#define B_SZ  32
#define H_SZ  512
#define NWG   32          // workgroups per direction
#define UH    16          // hidden units per workgroup
#define GH    64          // gate rows per workgroup (4 gates x 16 units)

typedef _Float16 half_t;
typedef __attribute__((ext_vector_type(8)))  _Float16 v8h;
typedef __attribute__((ext_vector_type(16))) _Float16 v16h;
typedef __attribute__((ext_vector_type(8)))  float    v8f;

// ---- workspace layout (bytes) ----
#define WS_CTRL 0              // dword0 = fwd barrier ctr, dword1 = bwd barrier ctr
#define WS_HBUF 256            // 2 dir * 2 phase * 32*512 halfs = 131072 B
#define WS_X16  131328         // T*B*H halfs (f16 embedded) = 16777216 B
#define WS_ZERO_DWORDS 32832   // ctrl(64 dw) + hbuf(32768 dw)

// ---- d_out sections (float elements) ----
#define OUT_HIDDEN 16777216    // after outputs [T,B,2,H]
#define OUT_EMB    33554432    // after hidden  [2,T,B,H]

// ---- LDS layout (bytes) ----
#define L_W     0              // 64 rows x 1032 halfs (padded stride) = 132096
#define L_X0    132096         // 32 rows x 520 halfs = 33280 (x ping)
#define L_X1    165376         // 32 rows x 520 halfs (x pong)
#define L_H     198656         // 32 rows x 520 halfs
#define L_GATES 231936         // 64 x 33 f32 = 8448
#define L_C     240384         // 16*32 f32
#define L_HPREV 242432         // 16*32 f32
#define L_BIAS  244480         // 64 f32
#define L_LEN   244736         // 32 i32
#define SMEM_BYTES 244864

__device__ __forceinline__ float sigf(float x) {
  return 1.0f / (1.0f + __expf(-x));
}
__device__ __forceinline__ float tanh_f(float x) {
  x = fminf(15.0f, fmaxf(-15.0f, x));
  float e = __expf(2.0f * x);
  return (e - 1.0f) / (e + 1.0f);
}

// Kernel 1: embedding gather (-> d_out fp32 + ws f16 copy) and zero the
// barrier counters + h ping-pong buffers (must happen every call).
__global__ __launch_bounds__(256) void embed_init_kernel(
    const int* __restrict__ seq, const float* __restrict__ emb,
    float* __restrict__ out, void* __restrict__ ws)
{
  unsigned gid = blockIdx.x * 256u + threadIdx.x;   // grid covers exactly T*B*H
  if (gid < WS_ZERO_DWORDS) ((unsigned*)ws)[gid] = 0u;

  unsigned t   = gid >> 14;            // /(B*H)
  unsigned rem = gid & 16383u;
  unsigned b   = rem >> 9;
  unsigned j   = rem & 511u;
  unsigned tok = (unsigned)seq[t * B_SZ + b];
  float v = emb[tok * H_SZ + j];
  out[OUT_EMB + gid] = v;
  half_t* x16 = (half_t*)((char*)ws + WS_X16);
  x16[gid] = (half_t)v;
}

// Kernel 2: persistent bidirectional LSTM. 64 WGs: blocks 0..31 forward,
// 32..63 backward. Each WG owns 16 hidden units (64 gate rows), keeps its
// f16 weight slice resident in LDS for all 512 steps, prefetches x[t+1]
// into a ping-pong LDS buffer during the elementwise/barrier phase, and
// syncs its direction-group once per step via a monotonic atomic counter.
__global__ __launch_bounds__(256, 1) void bilstm_persistent_kernel(
    const float* __restrict__ Wih_f, const float* __restrict__ Whh_f,
    const float* __restrict__ bih_f, const float* __restrict__ bhh_f,
    const float* __restrict__ Wih_b, const float* __restrict__ Whh_b,
    const float* __restrict__ bih_b, const float* __restrict__ bhh_b,
    const int* __restrict__ lens,
    float* __restrict__ out, void* __restrict__ ws)
{
  extern __shared__ char smem[];
  half_t* Wl = (half_t*)(smem + L_W);
  half_t* Hs = (half_t*)(smem + L_H);
  float*  Gl = (float*)(smem + L_GATES);
  float*  Cl = (float*)(smem + L_C);
  float*  Hp = (float*)(smem + L_HPREV);
  float*  Bl = (float*)(smem + L_BIAS);
  int*    Ll = (int*)(smem + L_LEN);

  const int tid = threadIdx.x;
  const int dir = blockIdx.x / NWG;       // 0 = forward, 1 = backward
  const int g   = blockIdx.x % NWG;
  const int j0  = g * UH;

  const float* Wih = dir ? Wih_b : Wih_f;
  const float* Whh = dir ? Whh_b : Whh_f;
  const float* bih = dir ? bih_b : bih_f;
  const float* bhh = dir ? bhh_b : bhh_f;

  unsigned* ctr = (unsigned*)ws + dir;
  half_t*       hbuf = (half_t*)((char*)ws + WS_HBUF);
  const half_t* x16  = (const half_t*)((char*)ws + WS_X16);

  // ---- prologue: stage fused weight slice [64 rows][K=1024] as f16 in LDS.
  // row r = q*16+u  <->  original gate row q*512 + j0 + u ; K<512 from Wih,
  // K>=512 from Whh. Padded row stride 1032 halfs (bank-conflict-free B gathers).
  for (int i = tid; i < GH * 1024; i += 256) {
    int r = i >> 10, k = i & 1023;
    int R = (r >> 4) * H_SZ + j0 + (r & 15);
    float w = (k < H_SZ) ? Wih[R * H_SZ + k] : Whh[R * H_SZ + (k - H_SZ)];
    Wl[r * 1032 + k] = (half_t)w;
  }
  if (tid < GH) {
    int R = (tid >> 4) * H_SZ + j0 + (tid & 15);
    Bl[tid] = bih[R] + bhh[R];
  }
  if (tid < B_SZ) Ll[tid] = lens[tid];
  for (int i = tid; i < UH * B_SZ; i += 256) { Cl[i] = 0.f; Hp[i] = 0.f; }

  // prefetch x[t0] into x ping buffer (used at s=0)
  {
    const int t0 = dir ? (T_LEN - 1) : 0;
    const half_t* xs0 = x16 + (size_t)t0 * (B_SZ * H_SZ);
    half_t* X0 = (half_t*)(smem + L_X0);
    for (int i = tid; i < 2048; i += 256) {
      int m = i >> 6, c = (i & 63) << 3;
      *(v8h*)(X0 + m * 520 + c) = *(const v8h*)(xs0 + m * 512 + c);
    }
  }
  __syncthreads();

  const int wave  = tid >> 5, lane = tid & 31;
  const int mtile = wave & 1;            // batch half (rows 0..15 / 16..31)
  const int qgate = wave >> 1;           // N-tile == gate index (i,f,g,o)
  const int lm = lane & 15, lh = lane >> 4;

  for (int s = 0; s < T_LEN; ++s) {
    const int t = dir ? (T_LEN - 1 - s) : s;
    const half_t* hrd = hbuf + (dir * 2 + (s & 1))       * (B_SZ * H_SZ);
    half_t*       hwr = hbuf + (dir * 2 + ((s + 1) & 1)) * (B_SZ * H_SZ);
    half_t* Xcur = (half_t*)(smem + ((s & 1) ? L_X1 : L_X0));

    // stage h (f16, fresh from the other WGPs via L2) into padded LDS rows.
    // This is the only global traffic on the step critical path.
    for (int i = tid; i < 2048; i += 256) {
      int m = i >> 6, c = (i & 63) << 3;
      *(v8h*)(Hs + m * 520 + c) = *(const v8h*)(hrd + m * 512 + c);
    }
    __syncthreads();

    // gates[32x64] = [x;h][32x1024] @ Wslice.T ; one 16x16 tile per wave.
    // Software-pipelined: fragment loads for chunk kc+1 are issued before the
    // wmma consuming chunk kc, so ds reads overlap the XDL pipe.
    v8f acc = {};
    const int arow = (mtile * 16 + lm) * 520;
    const int brow = (qgate * 16 + lm) * 1032;
    const half_t* Xc = Xcur + arow + lh * 8;   // A chunk kc: +kc*32 (halfs)
    const half_t* Hc = Hs   + arow + lh * 8;
    const half_t* Bp = Wl + brow + lh * 8;     // B chunk kc: +kc*32 (halfs)

    v8h alo = *(const v8h*)(Xc);
    v8h ahi = *(const v8h*)(Xc + 16);
    v8h blo = *(const v8h*)(Bp);
    v8h bhi = *(const v8h*)(Bp + 16);
    #pragma unroll
    for (int kc = 0; kc < 32; ++kc) {
      v16h a = __builtin_shufflevector(alo, ahi, 0,1,2,3,4,5,6,7,8,9,10,11,12,13,14,15);
      v16h b = __builtin_shufflevector(blo, bhi, 0,1,2,3,4,5,6,7,8,9,10,11,12,13,14,15);
      if (kc < 31) {
        const int kn = kc + 1;
        const half_t* an = (kn < 16) ? (Xc + kn * 32) : (Hc + (kn - 16) * 32);
        alo = *(const v8h*)(an);
        ahi = *(const v8h*)(an + 16);
        blo = *(const v8h*)(Bp + kn * 32);
        bhi = *(const v8h*)(Bp + kn * 32 + 16);
      }
      acc = __builtin_amdgcn_wmma_f32_16x16x32_f16(false, a, false, b,
                                                   (short)0, acc, false, false);
    }

    // scatter C tile: lane (0-15,16-31) = N=lm, M = lh*8 + v  (ISA C layout)
    #pragma unroll
    for (int v = 0; v < 8; ++v) {
      int m = mtile * 16 + lh * 8 + v;
      int n = qgate * 16 + lm;
      Gl[n * 33 + m] = acc[v];
    }

    // prefetch x[t_next] into the other x buffer; these global loads fly
    // during the elementwise phase and the inter-WGP barrier spin.
    if (s + 1 < T_LEN) {
      const int tn = dir ? (T_LEN - 2 - s) : (s + 1);
      const half_t* xsn = x16 + (size_t)tn * (B_SZ * H_SZ);
      half_t* Xn = (half_t*)(smem + ((s & 1) ? L_X0 : L_X1));
      for (int i = tid; i < 2048; i += 256) {
        int m = i >> 6, c = (i & 63) << 3;
        *(v8h*)(Xn + m * 520 + c) = *(const v8h*)(xsn + m * 512 + c);
      }
    }
    __syncthreads();

    // elementwise LSTM cell for this WG's 16 units x 32 batch (2 per thread)
    #pragma unroll
    for (int pp = 0; pp < 2; ++pp) {
      int p = tid + pp * 256;
      int m = p >> 4, u = p & 15;
      float gi = Gl[u * 33 + m]          + Bl[u];
      float gf = Gl[(16 + u) * 33 + m]   + Bl[16 + u];
      float gg = Gl[(32 + u) * 33 + m]   + Bl[32 + u];
      float go = Gl[(48 + u) * 33 + m]   + Bl[48 + u];
      float iv = sigf(gi), fv = sigf(gf), gv = tanh_f(gg), ov = sigf(go);
      int ci = u * B_SZ + m;
      float c_old = Cl[ci], h_old = Hp[ci];
      float c_new = fv * c_old + iv * gv;
      float h_new = ov * tanh_f(c_new);
      bool  msk = t < Ll[m];
      float c2 = msk ? c_new : c_old;
      float h2 = msk ? h_new : h_old;
      float o2 = msk ? h_new : 0.f;
      Cl[ci] = c2; Hp[ci] = h2;
      int jj = j0 + u;
      out[(size_t)t * (B_SZ * 2 * H_SZ) + m * (2 * H_SZ) + dir * H_SZ + jj] = o2;
      out[OUT_HIDDEN + (size_t)dir * (T_LEN * B_SZ * H_SZ)
                     + (size_t)t * (B_SZ * H_SZ) + m * H_SZ + jj] = o2;
      hwr[m * H_SZ + jj] = (half_t)h2;
    }

    // one device-scope barrier per step (ping-pong makes single barrier safe:
    // everyone READ buf[s&1] before this point, everyone WROTE buf[(s+1)&1])
    __threadfence();        // each wave: flush its h stores to device scope
    __syncthreads();
    if (tid == 0) {
      __hip_atomic_fetch_add(ctr, 1u, __ATOMIC_RELEASE, __HIP_MEMORY_SCOPE_AGENT);
      unsigned tgt = (unsigned)(s + 1) * NWG;
      while (__hip_atomic_load(ctr, __ATOMIC_ACQUIRE, __HIP_MEMORY_SCOPE_AGENT) < tgt)
        __builtin_amdgcn_s_sleep(1);
    }
    __syncthreads();
    __threadfence();        // each wave: invalidate so fresh h is visible
  }
}

extern "C" void kernel_launch(void* const* d_in, const int* in_sizes, int n_in,
                              void* d_out, int out_size, void* d_ws, size_t ws_size,
                              hipStream_t stream) {
  (void)in_sizes; (void)n_in; (void)out_size; (void)ws_size;
  const int*   seq   = (const int*)d_in[0];
  const int*   lens  = (const int*)d_in[1];
  const float* emb   = (const float*)d_in[2];
  const float* Wih_f = (const float*)d_in[3];
  const float* Whh_f = (const float*)d_in[4];
  const float* bih_f = (const float*)d_in[5];
  const float* bhh_f = (const float*)d_in[6];
  const float* Wih_b = (const float*)d_in[7];
  const float* Whh_b = (const float*)d_in[8];
  const float* bih_b = (const float*)d_in[9];
  const float* bhh_b = (const float*)d_in[10];
  float* out = (float*)d_out;

  // allow > 64KB dynamic LDS (WGP has 320KB); harmless if already set
  (void)hipFuncSetAttribute(reinterpret_cast<const void*>(bilstm_persistent_kernel),
                            hipFuncAttributeMaxDynamicSharedMemorySize, SMEM_BYTES);

  // T*B*H = 8,388,608 elements == 32768 blocks * 256 threads exactly
  embed_init_kernel<<<32768, 256, 0, stream>>>(seq, emb, out, d_ws);
  bilstm_persistent_kernel<<<2 * NWG, 256, SMEM_BYTES, stream>>>(
      Wih_f, Whh_f, bih_f, bhh_f, Wih_b, Whh_b, bih_b, bhh_b, lens, out, d_ws);
}
